// ScaledDotProductSelector_29884382446394
// MI455X (gfx1250) — matrix operand order repaired and checked by reading, not compiled, not run on hardware
//
#include <hip/hip_runtime.h>
#include <hip/hip_bf16.h>

typedef __attribute__((ext_vector_type(16))) _Float16 v16h;
typedef __attribute__((ext_vector_type(2)))  _Float16 h2;
typedef __attribute__((ext_vector_type(8)))  float    v8f;

#define BATCH 2
#define HEADS 16
#define SEQ   2048
#define DIM   64
#define QT    128   // query rows per workgroup (8 waves x 16 rows)
#define KT    64    // kv rows per LDS tile
#define PADH  72    // LDS row stride in halves (64 + 8 pad; 144B rows stay 16B-aligned)

__global__ __launch_bounds__(256)
void fa_causal_fwd(const float* __restrict__ q,
                   const float* __restrict__ k,
                   const float* __restrict__ v,
                   float* __restrict__ out)
{
    __shared__ _Float16 Qs[QT][PADH];        // 18.0 KB
    __shared__ _Float16 Ks[KT][PADH];        //  9.0 KB  row-major K tile
    __shared__ _Float16 Vt[DIM][PADH];       //  9.0 KB  TRANSPOSED V tile: Vt[d][kv]
    __shared__ _Float16 Ps[8][16][PADH];     // 18.0 KB  per-wave P scratch

    const int tid  = threadIdx.x;
    const int wid  = tid >> 5;               // wave id 0..7
    const int lane = tid & 31;
    const int m    = lane & 15;              // row for A-frags, col for B/C-frags
    const int hs   = lane >> 4;              // half-wave select

    const int bh   = blockIdx.y;             // fused batch*head
    const int q0   = blockIdx.x * QT;        // first query row of this block
    const size_t base = (size_t)bh * SEQ * DIM;

    const int wrow0  = q0 + wid * 16;        // first query row owned by this wave
    const int rowmax = wrow0 + 15;           // last  query row owned by this wave

    // ---- stage Q tile: f32 global -> f16 LDS ----
    {
        const float4* q4 = reinterpret_cast<const float4*>(q + base + (size_t)q0 * DIM);
        for (int i = tid; i < QT * (DIM / 4); i += 256) {
            int r = i >> 4, c = (i & 15) * 4;
            float4 val = q4[i];
            Qs[r][c + 0] = (_Float16)val.x;
            Qs[r][c + 1] = (_Float16)val.y;
            Qs[r][c + 2] = (_Float16)val.z;
            Qs[r][c + 3] = (_Float16)val.w;
        }
    }
    __syncthreads();

    // ---- Q A-fragments, built once, live in VGPRs for the whole KV loop ----
    v16h aQ[2];
    {
        const h2* qrow = reinterpret_cast<const h2*>(&Qs[wid * 16 + m][0]);
        #pragma unroll
        for (int s = 0; s < 2; ++s) {
            #pragma unroll
            for (int vv = 0; vv < 8; ++vv) {
                int kk = s * 32 + (vv >> 2) * 16 + hs * 8 + 2 * (vv & 3);
                h2 p = qrow[kk >> 1];
                aQ[s][2 * vv]     = p.x;
                aQ[s][2 * vv + 1] = p.y;
            }
        }
    }

    // all-ones B fragment (layout independent): row-sum of P via the matrix pipe
    v16h bOnes;
    #pragma unroll
    for (int i = 0; i < 16; ++i) bOnes[i] = (_Float16)1.0f;

    // ---- accumulators + online-softmax running max ----
    v8f Oacc[4];
    #pragma unroll
    for (int t = 0; t < 4; ++t) Oacc[t] = (v8f){};
    v8f Lacc = (v8f){};                      // row-sum accumulator (all cols identical)
    float mr[8];
    #pragma unroll
    for (int r = 0; r < 8; ++r) mr[r] = -1e30f;

    const int ntiles = 2 * (blockIdx.x + 1); // causal: kv tiles up to block diagonal

    for (int jt = 0; jt < ntiles; ++jt) {
        const int j0 = jt * KT;
        const bool active   = (j0 <= rowmax);              // any unmasked col in tile?
        const bool needMask = (j0 + KT - 1 > wrow0);       // tile crosses the diagonal?

        __syncthreads();   // prior iteration done reading Ks/Vt

        // ---- stage K (row-major) and V (transposed) tiles: f32 -> f16 LDS ----
        {
            const float4* k4 = reinterpret_cast<const float4*>(k + base + (size_t)j0 * DIM);
            const float4* v4 = reinterpret_cast<const float4*>(v + base + (size_t)j0 * DIM);
            for (int i = tid; i < KT * (DIM / 4); i += 256) {
                int r = i >> 4, c = (i & 15) * 4;
                float4 kv = k4[i];
                Ks[r][c + 0] = (_Float16)kv.x;
                Ks[r][c + 1] = (_Float16)kv.y;
                Ks[r][c + 2] = (_Float16)kv.z;
                Ks[r][c + 3] = (_Float16)kv.w;
                float4 vv = v4[i];
                Vt[c + 0][r] = (_Float16)vv.x;
                Vt[c + 1][r] = (_Float16)vv.y;
                Vt[c + 2][r] = (_Float16)vv.z;
                Vt[c + 3][r] = (_Float16)vv.w;
            }
            // prefetch next kv tile while we compute on this one
            if (jt + 1 < ntiles) {
                const char* pk = (const char*)(k + base + (size_t)(j0 + KT) * DIM) + tid * 64;
                const char* pv = (const char*)(v + base + (size_t)(j0 + KT) * DIM) + tid * 64;
                __builtin_prefetch(pk, 0, 1);
                __builtin_prefetch(pv, 0, 1);
            }
        }
        __syncthreads();

        v8f Sacc[4];
        if (active) {
            // ---- S = Q * K^T : 4 N-tiles x 2 k-steps ----
            #pragma unroll
            for (int t = 0; t < 4; ++t) {
                v8f acc = (v8f){};
                const h2* krow = reinterpret_cast<const h2*>(&Ks[t * 16 + m][0]);
                #pragma unroll
                for (int s = 0; s < 2; ++s) {
                    v16h bK;
                    #pragma unroll
                    for (int vv = 0; vv < 8; ++vv) {
                        int kk = s * 32 + (vv >> 2) * 16 + hs * 8 + 2 * (vv & 3);
                        h2 p = krow[kk >> 1];
                        bK[2 * vv]     = p.x;
                        bK[2 * vv + 1] = p.y;
                    }
                    acc = __builtin_amdgcn_wmma_f32_16x16x32_f16(
                            false, aQ[s], false, bK, (short)0, acc, false, false);
                }
                Sacc[t] = acc;
            }

            // ---- causal mask (diagonal tiles only) ----
            if (needMask) {
                #pragma unroll
                for (int r = 0; r < 8; ++r) {
                    const int mabs = wrow0 + r + hs * 8;
                    #pragma unroll
                    for (int t = 0; t < 4; ++t) {
                        int nabs = j0 + t * 16 + m;
                        if (nabs > mabs) Sacc[t][r] = -1e30f;
                    }
                }
            }

            // ---- online softmax: row max butterfly, rescale O & L ----
            #pragma unroll
            for (int r = 0; r < 8; ++r) {
                float mx = fmaxf(fmaxf(Sacc[0][r], Sacc[1][r]),
                                 fmaxf(Sacc[2][r], Sacc[3][r]));
                #pragma unroll
                for (int off = 1; off <= 8; off <<= 1)
                    mx = fmaxf(mx, __shfl_xor(mx, off, 32));
                float mnew  = fmaxf(mr[r], mx);
                float alpha = __expf(mr[r] - mnew);
                mr[r] = mnew;
                #pragma unroll
                for (int t = 0; t < 4; ++t) {
                    Sacc[t][r] = __expf(Sacc[t][r] - mnew);
                    Oacc[t][r] *= alpha;
                }
                Lacc[r] *= alpha;
            }

            // ---- spill P (f16) to per-wave LDS scratch ----
            #pragma unroll
            for (int t = 0; t < 4; ++t)
                #pragma unroll
                for (int r = 0; r < 8; ++r)
                    Ps[wid][r + hs * 8][t * 16 + m] = (_Float16)Sacc[t][r];
        }
        __syncthreads();   // P visible across lanes; Vt stable for reads below

        if (active) {
            // ---- O += P*V and L += P*1 : 2 k-steps x (4 d-tiles + 1 sum) ----
            const h2* prow = reinterpret_cast<const h2*>(&Ps[wid][m][0]);
            #pragma unroll
            for (int s = 0; s < 2; ++s) {
                v16h aP;
                #pragma unroll
                for (int vv = 0; vv < 8; ++vv) {
                    int kk = s * 32 + (vv >> 2) * 16 + hs * 8 + 2 * (vv & 3);
                    h2 p = prow[kk >> 1];
                    aP[2 * vv]     = p.x;
                    aP[2 * vv + 1] = p.y;
                }
                #pragma unroll
                for (int t = 0; t < 4; ++t) {
                    v16h bV;
                    const h2* vrow = reinterpret_cast<const h2*>(&Vt[t * 16 + m][0]);
                    #pragma unroll
                    for (int vv = 0; vv < 8; ++vv) {
                        int kk = s * 32 + (vv >> 2) * 16 + hs * 8 + 2 * (vv & 3);
                        h2 p = vrow[kk >> 1];
                        bV[2 * vv]     = p.x;
                        bV[2 * vv + 1] = p.y;
                    }
                    Oacc[t] = __builtin_amdgcn_wmma_f32_16x16x32_f16(
                                false, aP, false, bV, (short)0, Oacc[t], false, false);
                }
                Lacc = __builtin_amdgcn_wmma_f32_16x16x32_f16(
                         false, aP, false, bOnes, (short)0, Lacc, false, false);
            }
        }
    }

    // ---- epilogue: O /= rowsum, store fp32 ----
    float* orow = out + base;
    #pragma unroll
    for (int r = 0; r < 8; ++r) {
        float inv = 1.0f / Lacc[r];
        int row = wrow0 + r + hs * 8;
        #pragma unroll
        for (int t = 0; t < 4; ++t)
            orow[(size_t)row * DIM + t * 16 + m] = Oacc[t][r] * inv;
    }
}

extern "C" void kernel_launch(void* const* d_in, const int* in_sizes, int n_in,
                              void* d_out, int out_size, void* d_ws, size_t ws_size,
                              hipStream_t stream) {
    (void)in_sizes; (void)n_in; (void)out_size; (void)d_ws; (void)ws_size;
    const float* q = (const float*)d_in[0];
    const float* k = (const float*)d_in[1];
    const float* v = (const float*)d_in[2];
    float* out = (float*)d_out;
    dim3 grid(SEQ / QT, BATCH * HEADS);
    dim3 block(256);
    fa_causal_fwd<<<grid, block, 0, stream>>>(q, k, v, out);
}
